// GraphBlock_52467320488062
// MI455X (gfx1250) — compile-verified, matrix operand order checked by reading, not run on hardware
//
#include <hip/hip_runtime.h>
#include <hip/hip_bf16.h>

#define N_NODES  50000
#define N_EDGES  800000
#define D_IN     128
#define D_HID    256
#define D_OUT    128
#define N_GRAPHS 512
#define BN_EPS   1e-5f

typedef __bf16 bf16_t;
typedef __attribute__((ext_vector_type(16))) __bf16 v16bf;
typedef __attribute__((ext_vector_type(8)))  __bf16 v8bf;
typedef __attribute__((ext_vector_type(8)))  float  v8f;

// ---------------------------------------------------------------------------
// CDNA5 async global->LDS copy (ASYNCcnt-tracked, no VGPR data path).
// LDS dest address = low 32 bits of the generic LDS pointer (ISA 10.2).
// ---------------------------------------------------------------------------
__device__ __forceinline__ void async_copy16(void* lds_dst, const void* gsrc) {
    unsigned int       l = (unsigned int)(unsigned long long)(uintptr_t)lds_dst;
    unsigned long long g = (unsigned long long)(uintptr_t)gsrc;
    asm volatile("global_load_async_to_lds_b128 %0, %1, off"
                 :: "v"(l), "v"(g) : "memory");
}
__device__ __forceinline__ void wait_async0() {
    asm volatile("s_wait_asynccnt 0x0" ::: "memory");
}

// ---------------------------------------------------------------------------
// h = x  (GIN (1+eps)*x_i term, eps = 0)
// ---------------------------------------------------------------------------
__global__ void copy_x(const float* __restrict__ x, float* __restrict__ h) {
    const int n4 = (N_NODES * D_IN) / 4;
    for (int i = blockIdx.x * blockDim.x + threadIdx.x; i < n4;
         i += gridDim.x * blockDim.x) {
        ((float4*)h)[i] = ((const float4*)x)[i];
    }
}

// ---------------------------------------------------------------------------
// bf16 transposed weights: W1T[n][k] = bf16(W1[k][n]), W2T likewise.
// ---------------------------------------------------------------------------
__global__ void prep_weights(const float* __restrict__ W1,
                             const float* __restrict__ W2,
                             bf16_t* __restrict__ W1T,   // [D_HID][D_IN]
                             bf16_t* __restrict__ W2T) { // [D_OUT][D_HID]
    int idx = blockIdx.x * blockDim.x + threadIdx.x;   // 0 .. 32767
    if (idx < D_IN * D_HID) {
        int k = idx / D_HID, n = idx % D_HID;
        W1T[n * D_IN + k] = (bf16_t)W1[idx];
    }
    if (idx < D_HID * D_OUT) {
        int k = idx / D_OUT, n = idx % D_OUT;
        W2T[n * D_HID + k] = (bf16_t)W2[idx];
    }
}

// ---------------------------------------------------------------------------
// Edge scatter-add: h[dst] += x[src].  One wave32 per edge, 4 floats / lane.
// ---------------------------------------------------------------------------
__global__ void scatter_edges(const float* __restrict__ x,
                              const int* __restrict__ ei,
                              float* __restrict__ h) {
    int e = blockIdx.x * 8 + (threadIdx.x >> 5);
    if (e >= N_EDGES) return;
    int lane = threadIdx.x & 31;
    int src = ei[e];
    int dst = ei[N_EDGES + e];
    if ((unsigned)src >= N_NODES || (unsigned)dst >= N_NODES) return;
    const float* xs = x + (long)src * D_IN;
    float* hd = h + (long)dst * D_IN;
#pragma unroll
    for (int i = 0; i < 4; ++i)
        atomicAdd(&hd[lane + 32 * i], xs[lane + 32 * i]);
}

// ---------------------------------------------------------------------------
// Fragment loader: 8 contiguous bf16 at p, 8 more at p+16 (K-halves).
// ---------------------------------------------------------------------------
__device__ __forceinline__ v16bf load_bfrag(const bf16_t* p) {
    v8bf c0 = *(const v8bf*)p;
    v8bf c1 = *(const v8bf*)(p + 16);
    v16bf r;
#pragma unroll
    for (int i = 0; i < 8; ++i) { r[i] = c0[i]; r[8 + i] = c1[i]; }
    return r;
}

// ---------------------------------------------------------------------------
// Fused GIN MLP:  out = ReLU(h @ W1 + b1) @ W2 + b2
// 8 waves / block, 16-row tile per wave, bf16 WMMA, f32 accumulate.
// Weights streamed per 32-wide K-chunk into double-buffered LDS via
// global_load_async_to_lds_b128 (prefetch of chunk kc+1 overlaps WMMA of kc).
// ---------------------------------------------------------------------------
__launch_bounds__(256, 2)
__global__ void gin_mlp_wmma(const float*  __restrict__ h,
                             const bf16_t* __restrict__ W1T,  // [D_HID][D_IN]
                             const bf16_t* __restrict__ W2T,  // [D_OUT][D_HID]
                             const float*  __restrict__ b1,
                             const float*  __restrict__ b2,
                             float*        __restrict__ out) {
    // rows padded (136 / 40) to spread strided ds_load_b128 across banks
    __shared__ __attribute__((aligned(128))) bf16_t sW1[2][32][136];
    __shared__ __attribute__((aligned(128))) bf16_t sW2[2][128][40];
    __shared__ __attribute__((aligned(128))) bf16_t sHid[8][16][40];

    const int tid  = threadIdx.x;
    const int lane = tid & 31;
    const int wave = tid >> 5;
    const int l16  = lane & 15;
    const int half = lane >> 4;               // K-half / M-half selector
    const int rowBase = blockIdx.x * 128 + wave * 16;

    // stage K-chunk kc of both weight matrices into LDS buffer b (async)
    auto stage = [&](int kc, int b) {
        // W1T rows [kc*32, kc*32+32) x 128 K : 512 x 16B
#pragma unroll
        for (int i = 0; i < 2; ++i) {
            int c = tid + i * 256;
            int r = c >> 4, sub = c & 15;
            async_copy16(&sW1[b][r][sub * 8],
                         W1T + (kc * 32 + r) * D_IN + sub * 8);
        }
        // W2T all 128 rows x K slice [kc*32, kc*32+32) : 512 x 16B
#pragma unroll
        for (int i = 0; i < 2; ++i) {
            int c = tid + i * 256;
            int r = c >> 2, sub = c & 3;
            async_copy16(&sW2[b][r][sub * 8],
                         W2T + r * D_HID + kc * 32 + sub * 8);
        }
    };

    // ---- layer-1 A fragments: rows of h, f32 -> bf16 (clamped, no diverge) -
    int arow = rowBase + l16;
    if (arow >= N_NODES) arow = N_NODES - 1;
    const float* aptr = h + (long)arow * D_IN;

    v16bf a1[4];
#pragma unroll
    for (int kt = 0; kt < 4; ++kt) {
        const float4* p0 = (const float4*)(aptr + kt * 32 + half * 8);
        const float4* p1 = (const float4*)(aptr + kt * 32 + 16 + half * 8);
        float4 u0 = p0[0], u1 = p0[1], u2 = p1[0], u3 = p1[1];
        v16bf a;
        a[0]  = (bf16_t)u0.x; a[1]  = (bf16_t)u0.y; a[2]  = (bf16_t)u0.z; a[3]  = (bf16_t)u0.w;
        a[4]  = (bf16_t)u1.x; a[5]  = (bf16_t)u1.y; a[6]  = (bf16_t)u1.z; a[7]  = (bf16_t)u1.w;
        a[8]  = (bf16_t)u2.x; a[9]  = (bf16_t)u2.y; a[10] = (bf16_t)u2.z; a[11] = (bf16_t)u2.w;
        a[12] = (bf16_t)u3.x; a[13] = (bf16_t)u3.y; a[14] = (bf16_t)u3.z; a[15] = (bf16_t)u3.w;
        a1[kt] = a;
    }

    v8f acc2[8];
#pragma unroll
    for (int n = 0; n < 8; ++n) acc2[n] = (v8f){0.f,0.f,0.f,0.f,0.f,0.f,0.f,0.f};

    stage(0, 0);   // prefetch first chunk

    // ---- stream layer-2 K-chunks: each = two layer-1 N-tiles --------------
    for (int kc = 0; kc < 8; ++kc) {
        const int b = kc & 1;
        wait_async0();          // this wave's async writes to LDS landed
        __syncthreads();        // all waves' writes visible; prev compute done
        if (kc < 7) stage(kc + 1, b ^ 1);   // overlap prefetch with compute

#pragma unroll
        for (int s = 0; s < 2; ++s) {
            v8f acc1 = (v8f){0.f,0.f,0.f,0.f,0.f,0.f,0.f,0.f};
#pragma unroll
            for (int kt = 0; kt < 4; ++kt) {
                v16bf bb = load_bfrag(&sW1[b][s * 16 + l16][kt * 32 + half * 8]);
                acc1 = __builtin_amdgcn_wmma_f32_16x16x32_bf16(
                           false, a1[kt], false, bb, (short)0, acc1, false, false);
            }
            const float bias = b1[(kc * 2 + s) * 16 + l16];
#pragma unroll
            for (int r = 0; r < 8; ++r) {
                float v = acc1[r] + bias;
                v = v > 0.f ? v : 0.f;
                sHid[wave][half * 8 + r][s * 16 + l16] = (bf16_t)v;  // C -> A
            }
        }
        __syncthreads();

        // layer-2 A fragment from staged hidden chunk [16][32]
        v16bf a2 = load_bfrag(&sHid[wave][l16][half * 8]);

#pragma unroll
        for (int n2 = 0; n2 < 8; ++n2) {
            v16bf bb = load_bfrag(&sW2[b][n2 * 16 + l16][half * 8]);
            acc2[n2] = __builtin_amdgcn_wmma_f32_16x16x32_bf16(
                           false, a2, false, bb, (short)0, acc2[n2], false, false);
        }
        __syncthreads();
    }

    // ---- epilogue: + b2, store f32 ----------------------------------------
#pragma unroll
    for (int n2 = 0; n2 < 8; ++n2) {
        const float bias = b2[n2 * 16 + l16];
#pragma unroll
        for (int r = 0; r < 8; ++r) {
            int row = rowBase + half * 8 + r;
            if (row < N_NODES)
                out[(long)row * D_OUT + n2 * 16 + l16] = acc2[n2][r] + bias;
        }
    }
}

// ---------------------------------------------------------------------------
// zero scratch (pooled sums + counts)
// ---------------------------------------------------------------------------
__global__ void zero_f32(float* p, int n) {
    int i = blockIdx.x * blockDim.x + threadIdx.x;
    if (i < n) p[i] = 0.f;
}

// ---------------------------------------------------------------------------
// per-graph sum pool (atomics); one wave per node
// ---------------------------------------------------------------------------
__global__ void pool_accum(const float* __restrict__ hout,
                           const int* __restrict__ batch,
                           float* __restrict__ pooled,
                           float* __restrict__ counts) {
    int node = blockIdx.x * 8 + (threadIdx.x >> 5);
    if (node >= N_NODES) return;
    int lane = threadIdx.x & 31;
    int g = batch[node];
    if ((unsigned)g >= N_GRAPHS) return;
    const float* hp = hout + (long)node * D_OUT;
    float* pp = pooled + (long)g * D_OUT;
#pragma unroll
    for (int i = 0; i < 4; ++i)
        atomicAdd(&pp[lane + 32 * i], hp[lane + 32 * i]);
    if (lane == 0) atomicAdd(&counts[g], 1.0f);
}

// ---------------------------------------------------------------------------
// BatchNorm over graphs (biased var) + affine + ReLU.  One block per feature.
// ---------------------------------------------------------------------------
__global__ void bn_relu(const float* __restrict__ pooled,
                        const float* __restrict__ counts,
                        const float* __restrict__ gamma,
                        const float* __restrict__ beta,
                        float* __restrict__ out) {
    __shared__ float ssum[256];
    __shared__ float ssq[256];
    const int d = blockIdx.x;
    const int t = threadIdx.x;
    const int g0 = t, g1 = t + 256;

    float c0 = counts[g0]; c0 = c0 > 1.f ? c0 : 1.f;
    float c1 = counts[g1]; c1 = c1 > 1.f ? c1 : 1.f;
    float v0 = pooled[g0 * D_OUT + d] / c0;
    float v1 = pooled[g1 * D_OUT + d] / c1;

    ssum[t] = v0 + v1;
    ssq[t]  = v0 * v0 + v1 * v1;
    __syncthreads();
    for (int s = 128; s > 0; s >>= 1) {
        if (t < s) { ssum[t] += ssum[t + s]; ssq[t] += ssq[t + s]; }
        __syncthreads();
    }
    const float mean = ssum[0] * (1.0f / N_GRAPHS);
    const float var  = ssq[0] * (1.0f / N_GRAPHS) - mean * mean;
    const float inv  = rsqrtf(var + BN_EPS);
    const float gm = gamma[d], bt = beta[d];

    float o0 = (v0 - mean) * inv * gm + bt;
    float o1 = (v1 - mean) * inv * gm + bt;
    out[g0 * D_OUT + d] = o0 > 0.f ? o0 : 0.f;
    out[g1 * D_OUT + d] = o1 > 0.f ? o1 : 0.f;
}

// ---------------------------------------------------------------------------
extern "C" void kernel_launch(void* const* d_in, const int* in_sizes, int n_in,
                              void* d_out, int out_size, void* d_ws, size_t ws_size,
                              hipStream_t stream) {
    const float* x     = (const float*)d_in[0];
    const int*   ei    = (const int*)d_in[1];   // edge_index, flat [2*N_EDGES]
    const int*   batch = (const int*)d_in[2];
    const float* W1    = (const float*)d_in[3];
    const float* b1    = (const float*)d_in[4];
    const float* W2    = (const float*)d_in[5];
    const float* b2    = (const float*)d_in[6];
    const float* gamma = (const float*)d_in[7];
    const float* beta  = (const float*)d_in[8];
    float* out = (float*)d_out;

    // workspace layout (bytes)
    char* ws = (char*)d_ws;
    float*  h      = (float*)(ws);                       // 25,600,000
    float*  hout   = (float*)(ws + 25600000);            // 25,600,000
    bf16_t* W1T    = (bf16_t*)(ws + 51200000);           //     65,536
    bf16_t* W2T    = (bf16_t*)(ws + 51265536);           //     65,536
    float*  pooled = (float*)(ws + 51331072);            //    262,144
    float*  counts = (float*)(ws + 51593216);            //      2,048

    // 1) h = x
    copy_x<<<2048, 256, 0, stream>>>(x, h);
    // 2) bf16 transposed weights
    prep_weights<<<(D_IN * D_HID + 255) / 256, 256, 0, stream>>>(W1, W2, W1T, W2T);
    // 3) h += scatter_add(x[src] -> dst)
    scatter_edges<<<(N_EDGES + 7) / 8, 256, 0, stream>>>(x, ei, h);
    // 4) fused MLP (bf16 WMMA + async-LDS weight streaming)
    gin_mlp_wmma<<<(N_NODES + 127) / 128, 256, 0, stream>>>(h, W1T, W2T, b1, b2, hout);
    // 5) zero pool buffers, then per-graph sums
    zero_f32<<<(N_GRAPHS * D_OUT + N_GRAPHS + 255) / 256, 256, 0, stream>>>(pooled,
                                                        N_GRAPHS * D_OUT + N_GRAPHS);
    pool_accum<<<(N_NODES + 7) / 8, 256, 0, stream>>>(hout, batch, pooled, counts);
    // 6) BN + ReLU -> d_out
    bn_relu<<<D_OUT, 256, 0, stream>>>(pooled, counts, gamma, beta, out);

    (void)in_sizes; (void)n_in; (void)out_size; (void)ws_size;
}
